// GraphDecoder_31318901522605
// MI455X (gfx1250) — compile-verified, matrix operand order checked by reading, not run on hardware
//
#include <hip/hip_runtime.h>
#include <hip/hip_bf16.h>

// ---------------------------------------------------------------------------
// GraphDecoder forward for MI455X (gfx1250, wave32, WMMA).
// B=4, N=256, H=128, K=16, V=5000, E_DIM=64, L=64, TATTR=8, DVAL=16, GDN=2
// ---------------------------------------------------------------------------

#define Bsz 4
#define Nn  256
#define Hh  128
#define Kk  16
#define Vv  5000
#define Ee  64
#define Ll  64
#define TATTR 8
#define DVAL 16
#define ROWS (Bsz*Nn)          // 1024

typedef __attribute__((ext_vector_type(2)))  float  v2f;
typedef __attribute__((ext_vector_type(8)))  float  v8f;
typedef __attribute__((ext_vector_type(16))) __bf16 v16bf;

__device__ __forceinline__ float sigf(float x) { return 1.f / (1.f + __expf(-x)); }

// ---------------------------------------------------------------------------
// Kernel 1: node GRU chain (input-free GRU). One block per batch, sequential
// over N=256 steps. 512 threads: (hdim j = tid>>2, quarter q = tid&3),
// shuffle-reduced partial dots.
// ---------------------------------------------------------------------------
__global__ void k_node_chain(const float* __restrict__ latent,
                             const float* __restrict__ init_w,
                             const float* __restrict__ init_b,
                             const float* __restrict__ node_wh,
                             const float* __restrict__ node_bi,
                             const float* __restrict__ node_bh,
                             float* __restrict__ node_hiddens) {
  __shared__ float h[Hh];
  __shared__ float lat[Ll];
  const int b = blockIdx.x, tid = threadIdx.x;
  const int j = tid >> 2, q = tid & 3;
  if (tid < Ll) lat[tid] = latent[b*Ll + tid];
  __syncthreads();
  // h0 = relu(latent @ init_w.T + init_b)
  float s = 0.f;
  for (int k = q*16; k < q*16 + 16; ++k) s += lat[k] * init_w[j*Ll + k];
  s += __shfl_xor(s, 1); s += __shfl_xor(s, 2);
  if (q == 0) h[j] = fmaxf(s + init_b[j], 0.f);
  __syncthreads();
  for (int step = 0; step < Nn; ++step) {
    float sr = 0.f, sz = 0.f, sn = 0.f;
    for (int k = q*32; k < q*32 + 32; ++k) {
      float hv = h[k];
      sr += hv * node_wh[(      j)*Hh + k];
      sz += hv * node_wh[(Hh  + j)*Hh + k];
      sn += hv * node_wh[(2*Hh+ j)*Hh + k];
    }
    sr += __shfl_xor(sr, 1); sr += __shfl_xor(sr, 2);
    sz += __shfl_xor(sz, 1); sz += __shfl_xor(sz, 2);
    sn += __shfl_xor(sn, 1); sn += __shfl_xor(sn, 2);
    __syncthreads();
    if (q == 0) {
      float r  = sigf(node_bi[j]       + sr + node_bh[j]);
      float z  = sigf(node_bi[Hh + j]  + sz + node_bh[Hh + j]);
      float nn = tanhf(node_bi[2*Hh+j] + r * (sn + node_bh[2*Hh + j]));
      float hn = (1.f - z) * nn + z * h[j];
      h[j] = hn;
      node_hiddens[(b*Nn + step)*Hh + j] = hn;
    }
    __syncthreads();
  }
}

// ---------------------------------------------------------------------------
// Kernel 2: generic fp32 WMMA GEMM.  C[M,Nc] = A[M,K] @ W[Nc,K]^T + bias[n]
// One wave per 16x16 C tile.  grid = (M/16, Nc/16).  K multiple of 4.
// f32 A frag: a[v] = A[m][k0 + v + 2*half] ; B frag: b[v] = W[n][k0+v+2*half].
// ---------------------------------------------------------------------------
__global__ void k_gemm_f32(const float* __restrict__ A,
                           const float* __restrict__ W,
                           const float* __restrict__ bias,
                           float* __restrict__ C,
                           int K, int Nc) {
  const int mt = blockIdx.x, nt = blockIdx.y;
  const int lane = threadIdx.x & 31, half = lane >> 4, l15 = lane & 15;
  const int m = mt*16 + l15;
  const int n = nt*16 + l15;
  v8f acc = {};
  for (int k0 = 0; k0 < K; k0 += 4) {
    const float2 av = *(const float2*)(A + m*K + k0 + 2*half);
    const float2 bv = *(const float2*)(W + n*K + k0 + 2*half);
    v2f a; a[0] = av.x; a[1] = av.y;
    v2f b; b[0] = bv.x; b[1] = bv.y;
    acc = __builtin_amdgcn_wmma_f32_16x16x4_f32(false, a, false, b,
                                                (short)0, acc, false, false);
  }
  const float bn = bias ? bias[n] : 0.f;
  for (int v = 0; v < 8; ++v) {
    const int mr = mt*16 + v + 8*half;
    C[mr*Nc + n] = acc[v] + bn;
  }
}

// ---------------------------------------------------------------------------
// Kernel 3: fused scalar-input GRU chain (edge & value recurrences).
// Block = 128 threads (4 waves) owning 16 independent row-chains.
// Gates computed with fp32 WMMA (each wave covers 96 gate columns).
// mode 0: edge (255 steps, sigmoid head, writes edge_probs + A + deg + hasE)
// mode 1: value (16 steps, linear head, writes values)
// ---------------------------------------------------------------------------
__global__ void k_gru_chain(const float* __restrict__ h_init,
                            const float* __restrict__ wi,   // [384]
                            const float* __restrict__ wh,   // [384,128]
                            const float* __restrict__ bi,
                            const float* __restrict__ bh,
                            const float* __restrict__ head_w, // [128]
                            const float* __restrict__ head_b, // [1]
                            float* __restrict__ out,
                            float* __restrict__ Amat,
                            float* __restrict__ deg,
                            int* __restrict__ hasE,
                            int steps, int mode) {
  __shared__ float hs[16][Hh];
  __shared__ float G[16][3*Hh];
  __shared__ float pprev[16], pnew[16];
  const int tid = threadIdx.x;
  const int row0 = blockIdx.x * 16;
  for (int i = tid; i < 16*Hh; i += 128) hs[i >> 7][i & 127] = h_init[row0*Hh + i];
  if (tid < 16) pprev[tid] = 0.f;
  int degcnt = 0;
  __syncthreads();
  const int wave = tid >> 5, lane = tid & 31, half = lane >> 4, l15 = lane & 15;
  const float hb = head_b[0];
  for (int s = 0; s < steps; ++s) {
    // ---- gates: gh = h @ wh^T, wave covers cols [wave*96, wave*96+96)
    for (int nt = 0; nt < 6; ++nt) {
      const int col = wave*96 + nt*16 + l15;
      v8f acc = {};
      for (int k0 = 0; k0 < Hh; k0 += 4) {
        v2f a; a[0] = hs[l15][k0 + 2*half]; a[1] = hs[l15][k0 + 2*half + 1];
        const float2 bv = *(const float2*)(wh + col*Hh + k0 + 2*half);
        v2f b; b[0] = bv.x; b[1] = bv.y;
        acc = __builtin_amdgcn_wmma_f32_16x16x4_f32(false, a, false, b,
                                                    (short)0, acc, false, false);
      }
      const float add = bh[col], wic = wi[col], bic = bi[col];
      for (int v = 0; v < 8; ++v) {
        const int m = v + 8*half;
        float val = acc[v] + add;                 // gh (+bh)
        if (col < 2*Hh) val += pprev[m]*wic + bic; // +gi for r,z
        G[m][col] = val;
      }
    }
    __syncthreads();
    // ---- GRU combine: thread = hdim t, iterate its 16 rows
    {
      const int t = tid;
      const float wint = wi[2*Hh + t], bint = bi[2*Hh + t];
      for (int m = 0; m < 16; ++m) {
        const float r  = sigf(G[m][t]);
        const float z  = sigf(G[m][Hh + t]);
        const float nn = tanhf(pprev[m]*wint + bint + r * G[m][2*Hh + t]);
        hs[m][t] = (1.f - z)*nn + z*hs[m][t];
      }
    }
    __syncthreads();
    // ---- head
    if (tid < 16) {
      const int m = tid;
      float sd = 0.f;
      for (int k = 0; k < Hh; ++k) sd += hs[m][k] * head_w[k];
      const float x = sd + hb;
      const int grow = row0 + m;
      if (mode == 0) {
        const float p = sigf(x);
        pnew[m] = p;
        const int b = grow >> 8, tnode = grow & 255;
        if (s < tnode) {
          out[b*Nn*Nn + tnode*Nn + s] = p;         // edge_probs[b,t,i=s]
          if (p > 0.5f) { Amat[b*Nn*Nn + s*Nn + tnode] = 1.f; degcnt++; }
        }
      } else {
        pnew[m] = x;
        out[grow*DVAL + s] = x;                    // values[b,n,s]
      }
    }
    __syncthreads();
    if (tid < 16) pprev[tid] = pnew[tid];
    __syncthreads();
  }
  if (mode == 0 && tid < 16) {
    const int grow = row0 + tid;
    deg[grow] = (float)degcnt;
    if (degcnt > 0) atomicAdd(hasE + (grow >> 8), 1);
  }
}

// ---------------------------------------------------------------------------
// Kernel 4: sd = sqrt(deg)
// ---------------------------------------------------------------------------
__global__ void k_sd(const float* __restrict__ deg, float* __restrict__ sd) {
  const int i = blockIdx.x*blockDim.x + threadIdx.x;
  if (i < ROWS) sd[i] = sqrtf(deg[i]);
}

// ---------------------------------------------------------------------------
// Kernel 5: Wadj[b,i,t] = A[b,i,t]*sd[b,i]*sd[b,t]
// ---------------------------------------------------------------------------
__global__ void k_wadj(const float* __restrict__ A, const float* __restrict__ sd,
                       float* __restrict__ Wadj) {
  const int idx = blockIdx.x*blockDim.x + threadIdx.x;
  if (idx >= Bsz*Nn*Nn) return;
  const int b = idx >> 16, it = idx & 65535, i = it >> 8, t = it & 255;
  Wadj[idx] = A[idx] * sd[b*Nn + i] * sd[b*Nn + t];
}

// ---------------------------------------------------------------------------
// Kernel 6: GDN aggregation: out[b,t,h] = select(hasE,
//    relu(sum_i Wadj[b,i,t]*xl[b,i,h]) + gbias[h],  relu(xl[b,t,h]+gbias[h]))
// fp32 WMMA, A' = Wadj^T (per batch, K=256).  grid = (16, 8, B)
// ---------------------------------------------------------------------------
__global__ void k_gdn_agg(const float* __restrict__ Wadj,
                          const float* __restrict__ xl,
                          const float* __restrict__ gbias,
                          const int* __restrict__ hasE,
                          float* __restrict__ Xout) {
  const int mt = blockIdx.x, ntile = blockIdx.y, b = blockIdx.z;
  const int lane = threadIdx.x & 31, half = lane >> 4, l15 = lane & 15;
  const int t = mt*16 + l15;      // output node (A' row)
  const int n = ntile*16 + l15;   // hidden dim
  const float* Wb = Wadj + b*Nn*Nn;
  v8f acc = {};
  for (int k0 = 0; k0 < Nn; k0 += 4) {
    const int k = k0 + 2*half;
    v2f a; a[0] = Wb[(k    )*Nn + t]; a[1] = Wb[(k + 1)*Nn + t];
    v2f bb; bb[0] = xl[(b*Nn + k    )*Hh + n];
            bb[1] = xl[(b*Nn + k + 1)*Hh + n];
    acc = __builtin_amdgcn_wmma_f32_16x16x4_f32(false, a, false, bb,
                                                (short)0, acc, false, false);
  }
  const bool he = hasE[b] != 0;
  const float g = gbias[n];
  for (int v = 0; v < 8; ++v) {
    const int tr = mt*16 + v + 8*half;
    const int row = b*Nn + tr;
    const float alt = xl[row*Hh + n];
    Xout[row*Hh + n] = he ? (fmaxf(acc[v], 0.f) + g) : fmaxf(alt + g, 0.f);
  }
}

// ---------------------------------------------------------------------------
// Kernel 7: GRU combine for input-free name chain (gi = name_bi).
// G[row,384] already holds gh+bh (GEMM with bias=name_bh). Updates h in place.
// ---------------------------------------------------------------------------
__global__ void k_gru_combine(const float* __restrict__ G, float* __restrict__ h,
                              const float* __restrict__ bi) {
  const int idx = blockIdx.x*blockDim.x + threadIdx.x;
  if (idx >= ROWS*Hh) return;
  const int row = idx >> 7, t = idx & 127;
  const float r  = sigf(bi[t]        + G[row*3*Hh + t]);
  const float z  = sigf(bi[Hh + t]   + G[row*3*Hh + Hh + t]);
  const float nn = tanhf(bi[2*Hh+t] + r * G[row*3*Hh + 2*Hh + t]);
  h[idx] = (1.f - z)*nn + z*h[idx];
}

// ---------------------------------------------------------------------------
// Kernel 8: per-row 128-dot heads (stop_p / attr_dims)
// ---------------------------------------------------------------------------
__global__ void k_rowdot(const float* __restrict__ X, const float* __restrict__ w,
                         const float* __restrict__ bias, float* __restrict__ out,
                         int act) {
  const int row = blockIdx.x*blockDim.x + threadIdx.x;
  if (row >= ROWS) return;
  float s = bias[0];
  for (int k = 0; k < Hh; ++k) s += X[row*Hh + k] * w[k];
  out[row] = (act == 0) ? (1.f - sigf(s))                // stop: clip(1-sigmoid)
                        : logf(1.f + __expf(s));         // softplus
}

// ---------------------------------------------------------------------------
// Kernel 9: f32 -> bf16 (round-to-nearest-even), stored as raw u16 bits
// ---------------------------------------------------------------------------
__global__ void k_f2bf(const float* __restrict__ in, unsigned short* __restrict__ out,
                       int n) {
  const int i = blockIdx.x*blockDim.x + threadIdx.x;
  if (i >= n) return;
  const unsigned u = __float_as_uint(in[i]);
  out[i] = (unsigned short)((u + 0x7FFFu + ((u >> 16) & 1u)) >> 16);
}

// ---------------------------------------------------------------------------
// Kernel 10: name_logits = proj[8192,64] @ vocab[5000,64]^T  (bf16 WMMA, K=64)
// proj rows m = t*1024 + r ; output index ((r*8 + t)*5000 + n).
// Each wave keeps 4 accumulator tiles (64 output columns) and reuses one A
// fragment across them per K-step: 4x less A-fragment traffic, pushing the
// kernel toward the 23.3 TB/s store floor (164 MB fp32 output).
// bf16 A frag: a[v] packs k = 32*ki + 16*(v>>2) + 2*(v&3) + 8*half (and k+1).
// ---------------------------------------------------------------------------
__global__ void k_logits_bf16(const unsigned short* __restrict__ proj,
                              const unsigned short* __restrict__ voc,
                              float* __restrict__ out) {
  const int mt = blockIdx.x, ng = blockIdx.y;     // ng covers 4 n-tiles
  const int lane = threadIdx.x & 31, half = lane >> 4, l15 = lane & 15;
  const int m = mt*16 + l15;
  v8f acc0 = {}, acc1 = {}, acc2 = {}, acc3 = {};
  for (int ki = 0; ki < 2; ++ki) {
    union { v16bf v; unsigned u[8]; } ua;
    int kk[8];
    for (int vv = 0; vv < 8; ++vv) {
      kk[vv] = 32*ki + 16*(vv >> 2) + 2*(vv & 3) + 8*half;
      ua.u[vv] = *(const unsigned*)(proj + m*Ee + kk[vv]);
    }
    for (int j = 0; j < 4; ++j) {
      const int n = (ng*4 + j)*16 + l15;
      union { v16bf v; unsigned u[8]; } ub;
      for (int vv = 0; vv < 8; ++vv)
        ub.u[vv] = (n < Vv) ? *(const unsigned*)(voc + n*Ee + kk[vv]) : 0u;
      v8f* accp = (j == 0) ? &acc0 : (j == 1) ? &acc1 : (j == 2) ? &acc2 : &acc3;
      *accp = __builtin_amdgcn_wmma_f32_16x16x32_bf16(false, ua.v, false, ub.v,
                                                      (short)0, *accp, false, false);
    }
  }
  for (int j = 0; j < 4; ++j) {
    const v8f acc = (j == 0) ? acc0 : (j == 1) ? acc1 : (j == 2) ? acc2 : acc3;
    const int n = (ng*4 + j)*16 + l15;
    if (n >= Vv) continue;
    for (int vv = 0; vv < 8; ++vv) {
      const int mr = mt*16 + vv + 8*half;      // 0..8191
      const int t = mr >> 10, r = mr & 1023;
      out[(size_t)(r*TATTR + t)*Vv + n] = acc[vv];
    }
  }
}

// ---------------------------------------------------------------------------
// Host orchestration
// ---------------------------------------------------------------------------
extern "C" void kernel_launch(void* const* d_in, const int* in_sizes, int n_in,
                              void* d_out, int out_size, void* d_ws, size_t ws_size,
                              hipStream_t stream) {
  (void)in_sizes; (void)n_in; (void)out_size; (void)ws_size;
  const float* latent    = (const float*)d_in[0];
  const float* init_w    = (const float*)d_in[1];
  const float* init_b    = (const float*)d_in[2];
  const float* node_wh   = (const float*)d_in[3];
  const float* node_bi   = (const float*)d_in[4];
  const float* node_bh   = (const float*)d_in[5];
  const float* stop_w    = (const float*)d_in[6];
  const float* stop_b    = (const float*)d_in[7];
  const float* nodehead_w= (const float*)d_in[8];
  const float* nodehead_b= (const float*)d_in[9];
  const float* type_w    = (const float*)d_in[10];
  const float* type_b    = (const float*)d_in[11];
  const float* edge_wi   = (const float*)d_in[12];
  const float* edge_wh   = (const float*)d_in[13];
  const float* edge_bi   = (const float*)d_in[14];
  const float* edge_bh   = (const float*)d_in[15];
  const float* edgehead_w= (const float*)d_in[16];
  const float* edgehead_b= (const float*)d_in[17];
  const float* name_wh   = (const float*)d_in[19];
  const float* name_bi   = (const float*)d_in[20];
  const float* name_bh   = (const float*)d_in[21];
  const float* namehead_w= (const float*)d_in[22];
  const float* namehead_b= (const float*)d_in[23];
  const float* dims_w    = (const float*)d_in[24];
  const float* dims_b    = (const float*)d_in[25];
  const float* val_wi    = (const float*)d_in[26];
  const float* val_wh    = (const float*)d_in[27];
  const float* val_bi    = (const float*)d_in[28];
  const float* val_bh    = (const float*)d_in[29];
  const float* valhead_w = (const float*)d_in[30];
  const float* valhead_b = (const float*)d_in[31];
  const float* vocab_emb = (const float*)d_in[32];
  const float* gdn_w     = (const float*)d_in[33];
  const float* gdn_b     = (const float*)d_in[34];
  const float* gdn_bias  = (const float*)d_in[35];

  float* out = (float*)d_out;
  // output offsets (floats), in reference return order
  const size_t o_x    = 0;
  const size_t o_type = 131072;
  const size_t o_stop = 147456;
  const size_t o_edge = 148480;
  const size_t o_name = 410624;
  const size_t o_dims = 41370624;
  const size_t o_vals = 41371648;

  // workspace offsets (floats)
  float* ws = (float*)d_ws;
  const size_t OFF_NH   = 0;        // node_hiddens  [1024,128]
  const size_t OFF_NE   = 131072;   // node_emb      [1024,128]
  const size_t OFF_A    = 262144;   // adjacency     [B,256,256]
  const size_t OFF_WADJ = 524288;   // Wadj          [B,256,256]
  const size_t OFF_DEG  = 786432;   // deg           [1024]
  const size_t OFF_SD   = 787456;   // sqrt(deg)     [1024]
  const size_t OFF_HASE = 788480;   // int[4]
  const size_t OFF_XL   = 789504;   // gdn linear    [1024,128]
  const size_t OFF_X1   = 920576;   // gdn interm    [1024,128]
  const size_t OFF_HN   = 1051648;  // name h state  [1024,128]
  const size_t OFF_G    = 1182720;  // name gates    [1024,384]
  const size_t OFF_PROJ = 1575936;  // proj fp32     [8,1024,64]
  const size_t OFF_PBF  = 2100224;  // proj bf16 (as u16, 524288 of them)
  const size_t OFF_VBF  = 2362368;  // vocab bf16 (320000 u16)

  // --- per-call re-init (graph replays must be deterministic) ---
  hipMemsetAsync((char*)d_out + o_edge*4, 0, (size_t)Bsz*Nn*Nn*4, stream);
  hipMemsetAsync((char*)d_ws + OFF_A*4,   0, (size_t)Bsz*Nn*Nn*4, stream);
  hipMemsetAsync((char*)d_ws + OFF_HASE*4,0, 16, stream);

  // --- node chain -> node_hiddens ---
  k_node_chain<<<Bsz, 512, 0, stream>>>(latent, init_w, init_b, node_wh,
                                        node_bi, node_bh, ws + OFF_NH);
  // stop_p
  k_rowdot<<<4, 256, 0, stream>>>(ws + OFF_NH, stop_w, stop_b, out + o_stop, 0);
  // node_emb = nh @ nodehead_w^T + b
  k_gemm_f32<<<dim3(ROWS/16, Hh/16), 32, 0, stream>>>(ws + OFF_NH, nodehead_w,
                                                      nodehead_b, ws + OFF_NE, Hh, Hh);
  // type_logits
  k_gemm_f32<<<dim3(ROWS/16, Kk/16), 32, 0, stream>>>(ws + OFF_NE, type_w,
                                                      type_b, out + o_type, Hh, Kk);
  // --- edge recurrence (255 steps) ---
  k_gru_chain<<<ROWS/16, 128, 0, stream>>>(ws + OFF_NH, edge_wi, edge_wh, edge_bi,
      edge_bh, edgehead_w, edgehead_b, out + o_edge, ws + OFF_A, ws + OFF_DEG,
      (int*)(ws + OFF_HASE), Nn - 1, 0);
  // sd, Wadj
  k_sd<<<4, 256, 0, stream>>>(ws + OFF_DEG, ws + OFF_SD);
  k_wadj<<<(Bsz*Nn*Nn)/256, 256, 0, stream>>>(ws + OFF_A, ws + OFF_SD, ws + OFF_WADJ);

  // --- GDN (2 layers) ---
  // layer 0: x0 = node_emb -> x1
  k_gemm_f32<<<dim3(ROWS/16, Hh/16), 32, 0, stream>>>(ws + OFF_NE, gdn_w + 0*Hh*Hh,
                                                      gdn_b + 0*Hh, ws + OFF_XL, Hh, Hh);
  k_gdn_agg<<<dim3(Nn/16, Hh/16, Bsz), 32, 0, stream>>>(ws + OFF_WADJ, ws + OFF_XL,
      gdn_bias + 0*Hh, (int*)(ws + OFF_HASE), ws + OFF_X1);
  // layer 1: x1 -> x (output slot 0)
  k_gemm_f32<<<dim3(ROWS/16, Hh/16), 32, 0, stream>>>(ws + OFF_X1, gdn_w + 1*Hh*Hh,
                                                      gdn_b + 1*Hh, ws + OFF_XL, Hh, Hh);
  k_gdn_agg<<<dim3(Nn/16, Hh/16, Bsz), 32, 0, stream>>>(ws + OFF_WADJ, ws + OFF_XL,
      gdn_bias + 1*Hh, (int*)(ws + OFF_HASE), out + o_x);

  // attr_dims (softplus head on refined emb)
  k_rowdot<<<4, 256, 0, stream>>>(out + o_x, dims_w, dims_b, out + o_dims, 1);

  // --- name recurrence (8 steps) on refined emb ---
  hipMemcpyAsync(ws + OFF_HN, out + o_x, (size_t)ROWS*Hh*4,
                 hipMemcpyDeviceToDevice, stream);
  for (int t = 0; t < TATTR; ++t) {
    k_gemm_f32<<<dim3(ROWS/16, (3*Hh)/16), 32, 0, stream>>>(ws + OFF_HN, name_wh,
                                                            name_bh, ws + OFF_G, Hh, 3*Hh);
    k_gru_combine<<<(ROWS*Hh)/256, 256, 0, stream>>>(ws + OFF_G, ws + OFF_HN, name_bi);
    k_gemm_f32<<<dim3(ROWS/16, Ee/16), 32, 0, stream>>>(ws + OFF_HN, namehead_w,
        namehead_b, ws + OFF_PROJ + (size_t)t*ROWS*Ee, Hh, Ee);
  }
  // bf16 casts + big logits GEMM -> name_logits [B,N,TATTR,V]
  k_f2bf<<<(TATTR*ROWS*Ee)/256, 256, 0, stream>>>(ws + OFF_PROJ,
      (unsigned short*)(ws + OFF_PBF), TATTR*ROWS*Ee);
  k_f2bf<<<(Vv*Ee + 255)/256, 256, 0, stream>>>(vocab_emb,
      (unsigned short*)(ws + OFF_VBF), Vv*Ee);
  k_logits_bf16<<<dim3((TATTR*ROWS)/16, (Vv + 63)/64), 32, 0, stream>>>(
      (unsigned short*)(ws + OFF_PBF), (unsigned short*)(ws + OFF_VBF), out + o_name);

  // --- value recurrence (16 steps, linear head) ---
  k_gru_chain<<<ROWS/16, 128, 0, stream>>>(out + o_x, val_wi, val_wh, val_bi,
      val_bh, valhead_w, valhead_b, out + o_vals, nullptr, nullptr,
      (int*)(ws + OFF_HASE), DVAL, 1);
}
